// GGNNModel_22033182229079
// MI455X (gfx1250) — compile-verified, matrix operand order checked by reading, not run on hardware
//
#include <hip/hip_runtime.h>
#include <cstdint>
#include <cstddef>

typedef _Float16 f16;
typedef __attribute__((ext_vector_type(8)))  _Float16 v8h;
typedef __attribute__((ext_vector_type(16))) _Float16 v16h;
typedef __attribute__((ext_vector_type(8)))  float    v8f;

#define EPSV 1e-5f
#define WPB 8   // waves per block for WMMA kernels (256 threads)

// ---------------------------------------------------------------- helpers
static __device__ __forceinline__ v16h cat8(v8h a, v8h b) {
  return __builtin_shufflevector(a, b, 0,1,2,3,4,5,6,7,8,9,10,11,12,13,14,15);
}

// A-fragment (16x32 f16, one K-chunk k0 in {0,32}) from a row-major f16 row.
// lane l: row = l&15 (caller passes that row's pointer), hi = l>>4.
// halves j<8  -> K = k0 + hi*8 + j
// halves j>=8 -> K = k0 + 16 + hi*8 + (j-8)
static __device__ __forceinline__ v16h afrag_f16(const f16* rowp, int k0, int hi) {
  const f16* p = rowp + k0 + hi * 8;
  v8h lo = *(const v8h*)p;
  v8h hh = *(const v8h*)(p + 16);
  return cat8(lo, hh);
}

static __device__ __forceinline__ v16h afrag_f32(const float* rowp, int k0, int hi) {
  const float* p = rowp + k0 + hi * 8;
  v8h lo, hh;
#pragma unroll
  for (int j = 0; j < 8; ++j) { lo[j] = (f16)p[j]; hh[j] = (f16)p[16 + j]; }
  return cat8(lo, hh);
}

// B-fragment from pre-swizzled buffer: layout [frag][lane][16 halves]
// (works for both global and LDS source pointers)
static __device__ __forceinline__ v16h bfrag(const f16* w, int f, int lane) {
  const f16* p = w + ((size_t)(f * 32 + lane)) * 16;
  return cat8(*(const v8h*)p, *(const v8h*)(p + 8));
}

static __device__ __forceinline__ v8f wmma(v16h a, v16h b, v8f c) {
  return __builtin_amdgcn_wmma_f32_16x16x32_f16(false, a, false, b, (short)0, c, false, false);
}

static __device__ __forceinline__ void lds_fence_wave() {
  asm volatile("s_wait_dscnt 0x0" ::: "memory");
}

static __device__ __forceinline__ float sigmoidf_(float x) {
  return 1.0f / (1.0f + __expf(-x));
}

static __device__ __forceinline__ void atomAddF(float* p, float v) {
  unsafeAtomicAdd(p, v);   // -> global_atomic_add_f32
}

// ------------------------------------------------- weight swizzle (f32 -> f16 B-frags)
// W row-major [K x 64].  frag f = kt*4 + nt, kt = f>>2.
__global__ void k_swizzle(const float* __restrict__ W, f16* __restrict__ out, int nfrag) {
  int t = blockIdx.x * blockDim.x + threadIdx.x;
  if (t >= nfrag * 32) return;
  int f = t >> 5, l = t & 31;
  int kt = f >> 2, nt = f & 3;
  int n  = nt * 16 + (l & 15);
  int hi = l >> 4;
  f16* o = out + (size_t)t * 16;
#pragma unroll
  for (int j = 0; j < 16; ++j) {
    int kk = kt * 32 + ((j < 8) ? (hi * 8 + j) : (16 + hi * 8 + (j - 8)));
    o[j] = (f16)W[kk * 64 + n];
  }
}

// ------------------------------------------------- degree / dis
__global__ void k_deg(const int* __restrict__ ei, float* __restrict__ deg, int E_) {
  int e = blockIdx.x * blockDim.x + threadIdx.x;
  if (e >= E_) return;
  atomAddF(&deg[ei[E_ + e]], 1.0f);
}
__global__ void k_dis(float* __restrict__ dis, int N_) {
  int i = blockIdx.x * blockDim.x + threadIdx.x;
  if (i >= N_) return;
  dis[i] = rsqrtf(dis[i] + 1.0f);   // +1 self loop
}

// ------------------------------------------------- small VALU GEMMs
__global__ void k_pre(const float* __restrict__ x, const float* __restrict__ W,
                      const float* __restrict__ b, float* __restrict__ out, int N_) {
  int g = blockIdx.x * blockDim.x + threadIdx.x;
  if (g >= N_ * 32) return;
  int node = g >> 5, c = g & 31;
  float s = b[c];
#pragma unroll
  for (int k = 0; k < 16; ++k) s += x[node * 16 + k] * W[k * 32 + c];
  out[g] = s;
}

__global__ void k_init(const float* __restrict__ hpre, const float* __restrict__ W,
                       const float* __restrict__ b, float* __restrict__ out, int N_) {
  int g = blockIdx.x * blockDim.x + threadIdx.x;
  if (g >= N_ * 64) return;
  int node = g >> 6, c = g & 63;
  float s = b[c];
#pragma unroll
  for (int k = 0; k < 32; ++k) s += hpre[node * 32 + k] * W[k * 64 + c];
  out[g] = s;
}

__global__ void k_fc2(const float* __restrict__ h, const float* __restrict__ W,
                      const float* __restrict__ b, float* __restrict__ out, int N_) {
  int node = blockIdx.x * blockDim.x + threadIdx.x;
  if (node >= N_) return;
  float s0 = b[0], s1 = b[1];
#pragma unroll
  for (int k = 0; k < 64; ++k) {
    float v = h[node * 64 + k];
    s0 += v * W[k * 2 + 0];
    s1 += v * W[k * 2 + 1];
  }
  out[node * 2 + 0] = s0;
  out[node * 2 + 1] = s1;
}

// ------------------------------------------------- GraphNorm
// scratch[0..C-1] = sum, scratch[C..2C-1] = sumsq  (zeroed before launch)
__global__ void k_gn_reduce(const float* __restrict__ x, float* __restrict__ scratch,
                            int N_, int C) {
  int g = blockIdx.x * blockDim.x + threadIdx.x;
  int total = gridDim.x * blockDim.x;
  int c = g % C;
  float s = 0.f, s2 = 0.f;
  for (int r = g / C; r < N_; r += total / C) {
    float v = x[(size_t)r * C + c];
    s += v; s2 += v * v;
  }
  atomAddF(&scratch[c], s);
  atomAddF(&scratch[C + c], s2);
}

__global__ void k_gn_apply(const float* __restrict__ x, const float* __restrict__ scratch,
                           const float* __restrict__ w, const float* __restrict__ b,
                           const float* __restrict__ ms,
                           float* __restrict__ outF, f16* __restrict__ outH,
                           int N_, int C, int doRelu) {
  int g = blockIdx.x * blockDim.x + threadIdx.x;
  if (g >= N_ * C) return;
  int c = g % C;
  float sum = scratch[c], sum2 = scratch[C + c];
  float inv = 1.0f / (float)N_;
  float mean = sum * inv;
  float mm = ms[c] * mean;
  float var = (sum2 - 2.0f * mm * sum + (float)N_ * mm * mm) * inv;
  float istd = rsqrtf(var + EPSV);
  float y = w[c] * (x[g] - mm) * istd + b[c];
  if (doRelu) y = fmaxf(y, 0.0f);
  outF[g] = y;
  if (outH) outH[g] = (f16)y;
}

// ------------------------------------------------- edge MLP + scatter (the hot kernel)
// Weights (B1a|B1b|B2 = 24KB) staged into LDS once per block; B-fragments then
// come from the DS pipe, leaving VMEM for gathers + atomics.
__global__ void __launch_bounds__(256)
k_edge(const int* __restrict__ ei, const float* __restrict__ ea,
       const float* __restrict__ dis, const f16* __restrict__ h16,
       float* __restrict__ aggr,
       const f16* __restrict__ B1a, const f16* __restrict__ B1b, const f16* __restrict__ B2,
       const float* __restrict__ w1e, const float* __restrict__ gb1,
       const float* __restrict__ gb2, const float* __restrict__ gw3,
       const float* __restrict__ gb3, int E_, int N_) {
  __shared__ __align__(16) f16 wlds[3 * 4096];        // 24KB: B1a, B1b, B2
  __shared__ __align__(16) f16 lds_t[WPB][16 * 64];   // 16KB activation staging

  // cooperative weight staging (each array = 512 x v8h chunks of 16B)
  {
    int t = threadIdx.x;
    const v8h* s0 = (const v8h*)B1a;
    const v8h* s1 = (const v8h*)B1b;
    const v8h* s2 = (const v8h*)B2;
    v8h* dst = (v8h*)wlds;
#pragma unroll
    for (int i = 0; i < 2; ++i) {
      int idx = t + i * 256;
      dst[idx]        = s0[idx];
      dst[512 + idx]  = s1[idx];
      dst[1024 + idx] = s2[idx];
    }
  }
  __syncthreads();
  const f16* L1a = wlds;
  const f16* L1b = wlds + 4096;
  const f16* L2  = wlds + 8192;

  int wave = threadIdx.x >> 5;
  int lane = threadIdx.x & 31;
  int tile = blockIdx.x * WPB + wave;
  int ntiles = (E_ + N_) >> 4;
  if (tile < ntiles) {                       // wave-uniform: EXEC stays all-ones
    int m = lane & 15, hi = lane >> 4;
    int e = tile * 16 + m;

    int row, col;
    bool selftile = (tile * 16 >= E_);       // tiles are uniformly real or self-loop
    if (!selftile) { row = ei[e]; col = ei[E_ + e]; }
    else           { row = e - E_; col = row; }

    // ea values for the 8 output rows this half-lane owns in C-layout
    float ea8[8];
    if (!selftile) {
      int eb = tile * 16 + hi * 8;
#pragma unroll
      for (int r = 0; r < 8; ++r) ea8[r] = ea[eb + r];
    } else {
#pragma unroll
      for (int r = 0; r < 8; ++r) ea8[r] = 1.0f;
    }

    const f16* rj = h16 + (size_t)row * 64;  // x_j
    const f16* ri = h16 + (size_t)col * 64;  // x_i
    v16h aj0 = afrag_f16(rj, 0, hi), aj1 = afrag_f16(rj, 32, hi);
    v16h ai0 = afrag_f16(ri, 0, hi), ai1 = afrag_f16(ri, 32, hi);

    f16* t1 = lds_t[wave];

    // ---- layer 1: relu(x_i@W1a + x_j@W1b + ea*w1e + b1) -> t1 (LDS, row-major f16)
#pragma unroll
    for (int nt = 0; nt < 4; ++nt) {
      int n = nt * 16 + m;
      float w1en = w1e[n], b1n = gb1[n];
      v8f acc;
#pragma unroll
      for (int r = 0; r < 8; ++r) acc[r] = ea8[r] * w1en + b1n;
      acc = wmma(ai0, bfrag(L1a, 0 * 4 + nt, lane), acc);
      acc = wmma(ai1, bfrag(L1a, 1 * 4 + nt, lane), acc);
      acc = wmma(aj0, bfrag(L1b, 0 * 4 + nt, lane), acc);
      acc = wmma(aj1, bfrag(L1b, 1 * 4 + nt, lane), acc);
#pragma unroll
      for (int r = 0; r < 8; ++r)
        t1[(r + 8 * hi) * 64 + n] = (f16)fmaxf(acc[r], 0.0f);
    }
    lds_fence_wave();
    v16h at0 = afrag_f16(t1 + m * 64, 0, hi);
    v16h at1 = afrag_f16(t1 + m * 64, 32, hi);
    lds_fence_wave();

    // ---- layer 2: relu(t1@W2 + b2) -> t1 (overwrite)
#pragma unroll
    for (int nt = 0; nt < 4; ++nt) {
      int n = nt * 16 + m;
      v8f acc;
#pragma unroll
      for (int r = 0; r < 8; ++r) acc[r] = gb2[n];
      acc = wmma(at0, bfrag(L2, 0 * 4 + nt, lane), acc);
      acc = wmma(at1, bfrag(L2, 1 * 4 + nt, lane), acc);
#pragma unroll
      for (int r = 0; r < 8; ++r)
        t1[(r + 8 * hi) * 64 + n] = (f16)fmaxf(acc[r], 0.0f);
    }
    lds_fence_wave();

    // ---- layer 3: gate = sigmoid(t2 . w3 + b3)   (per-edge scalar, both half-lanes)
    float acc3 = gb3[0];
#pragma unroll
    for (int kc = 0; kc < 8; ++kc) {
      v8h tv = *(const v8h*)(t1 + m * 64 + kc * 8);
#pragma unroll
      for (int j = 0; j < 8; ++j) acc3 += (float)tv[j] * gw3[kc * 8 + j];
    }
    float gate = sigmoidf_(acc3);
    float s = gate * dis[row] * dis[col];

    // ---- scatter: msg = s * x_j, channels already in this lane's A-fragments
    float* dst = aggr + (size_t)col * 64;
#pragma unroll
    for (int j = 0; j < 8; ++j) atomAddF(dst + hi * 8 + j,      s * (float)aj0[j]);
#pragma unroll
    for (int j = 0; j < 8; ++j) atomAddF(dst + 16 + hi * 8 + j, s * (float)aj0[j + 8]);
#pragma unroll
    for (int j = 0; j < 8; ++j) atomAddF(dst + 32 + hi * 8 + j, s * (float)aj1[j]);
#pragma unroll
    for (int j = 0; j < 8; ++j) atomAddF(dst + 48 + hi * 8 + j, s * (float)aj1[j + 8]);
  }
}

// ------------------------------------------------- GRU node update (writes hnew into aggr)
__global__ void __launch_bounds__(256)
k_node(const f16* __restrict__ h16, const float* __restrict__ h, float* __restrict__ aggr,
       const f16* __restrict__ Bz, const f16* __restrict__ Br, const f16* __restrict__ Bh,
       const float* __restrict__ zb, const float* __restrict__ rb,
       const float* __restrict__ hb, int N_) {
  __shared__ __align__(16) f16 lds_t[WPB][16 * 64];
  int wave = threadIdx.x >> 5;
  int lane = threadIdx.x & 31;
  int tile = blockIdx.x * WPB + wave;
  if (tile >= (N_ >> 4)) return;
  int m = lane & 15, hi = lane >> 4;
  int base = tile * 16;

  const f16*   hr = h16  + (size_t)(base + m) * 64;
  const float* gr = aggr + (size_t)(base + m) * 64;
  v16h ah0 = afrag_f16(hr, 0, hi), ah1 = afrag_f16(hr, 32, hi);
  v16h ag0 = afrag_f32(gr, 0, hi), ag1 = afrag_f32(gr, 32, hi);

  f16* rh = lds_t[wave];

  // ---- r = sigmoid([h,aggr]@Wr + rb);  rh = r*h -> LDS f16
#pragma unroll
  for (int nt = 0; nt < 4; ++nt) {
    int n = nt * 16 + m;
    v8f acc;
#pragma unroll
    for (int r = 0; r < 8; ++r) acc[r] = rb[n];
    acc = wmma(ah0, bfrag(Br, 0 * 4 + nt, lane), acc);
    acc = wmma(ah1, bfrag(Br, 1 * 4 + nt, lane), acc);
    acc = wmma(ag0, bfrag(Br, 2 * 4 + nt, lane), acc);
    acc = wmma(ag1, bfrag(Br, 3 * 4 + nt, lane), acc);
#pragma unroll
    for (int r = 0; r < 8; ++r) {
      int mm = r + 8 * hi;
      float rv = sigmoidf_(acc[r]);
      float hv = h[(size_t)(base + mm) * 64 + n];
      rh[mm * 64 + n] = (f16)(rv * hv);
    }
  }
  lds_fence_wave();
  v16h arh0 = afrag_f16(rh + m * 64, 0, hi);
  v16h arh1 = afrag_f16(rh + m * 64, 32, hi);
  lds_fence_wave();

  // ---- hc = relu([r*h, aggr]@Wh + hb)
  v8f hcacc[4];
#pragma unroll
  for (int nt = 0; nt < 4; ++nt) {
    int n = nt * 16 + m;
    v8f acc;
#pragma unroll
    for (int r = 0; r < 8; ++r) acc[r] = hb[n];
    acc = wmma(arh0, bfrag(Bh, 0 * 4 + nt, lane), acc);
    acc = wmma(arh1, bfrag(Bh, 1 * 4 + nt, lane), acc);
    acc = wmma(ag0,  bfrag(Bh, 2 * 4 + nt, lane), acc);
    acc = wmma(ag1,  bfrag(Bh, 3 * 4 + nt, lane), acc);
#pragma unroll
    for (int r = 0; r < 8; ++r) acc[r] = fmaxf(acc[r], 0.0f);
    hcacc[nt] = acc;
  }

  // ---- z = sigmoid([h,aggr]@Wz + zb);  hnew = (1-z)h + z*hc  -> aggr (in place)
#pragma unroll
  for (int nt = 0; nt < 4; ++nt) {
    int n = nt * 16 + m;
    v8f acc;
#pragma unroll
    for (int r = 0; r < 8; ++r) acc[r] = zb[n];
    acc = wmma(ah0, bfrag(Bz, 0 * 4 + nt, lane), acc);
    acc = wmma(ah1, bfrag(Bz, 1 * 4 + nt, lane), acc);
    acc = wmma(ag0, bfrag(Bz, 2 * 4 + nt, lane), acc);
    acc = wmma(ag1, bfrag(Bz, 3 * 4 + nt, lane), acc);
#pragma unroll
    for (int r = 0; r < 8; ++r) {
      int mm = r + 8 * hi;
      float z  = sigmoidf_(acc[r]);
      float hv = h[(size_t)(base + mm) * 64 + n];
      aggr[(size_t)(base + mm) * 64 + n] = (1.0f - z) * hv + z * hcacc[nt][r];
    }
  }
}

// ------------------------------------------------- fc1: [h0, h] @ W + b -> out (f32)
__global__ void __launch_bounds__(256)
k_fc1(const f16* __restrict__ h016, const f16* __restrict__ h16,
      float* __restrict__ out, const f16* __restrict__ Bf,
      const float* __restrict__ fb, int N_) {
  int wave = threadIdx.x >> 5;
  int lane = threadIdx.x & 31;
  int tile = blockIdx.x * WPB + wave;
  if (tile >= (N_ >> 4)) return;
  int m = lane & 15, hi = lane >> 4;
  int base = tile * 16;

  const f16* r0 = h016 + (size_t)(base + m) * 64;
  const f16* r1 = h16  + (size_t)(base + m) * 64;
  v16h a0 = afrag_f16(r0, 0, hi), a1 = afrag_f16(r0, 32, hi);
  v16h a2 = afrag_f16(r1, 0, hi), a3 = afrag_f16(r1, 32, hi);

#pragma unroll
  for (int nt = 0; nt < 4; ++nt) {
    int n = nt * 16 + m;
    v8f acc;
#pragma unroll
    for (int r = 0; r < 8; ++r) acc[r] = fb[n];
    acc = wmma(a0, bfrag(Bf, 0 * 4 + nt, lane), acc);
    acc = wmma(a1, bfrag(Bf, 1 * 4 + nt, lane), acc);
    acc = wmma(a2, bfrag(Bf, 2 * 4 + nt, lane), acc);
    acc = wmma(a3, bfrag(Bf, 3 * 4 + nt, lane), acc);
#pragma unroll
    for (int r = 0; r < 8; ++r)
      out[(size_t)(base + r + 8 * hi) * 64 + n] = acc[r];
  }
}

// =================================================================== host
static inline int cdiv(int a, int b) { return (a + b - 1) / b; }

extern "C" void kernel_launch(void* const* d_in, const int* in_sizes, int n_in,
                              void* d_out, int out_size, void* d_ws, size_t ws_size,
                              hipStream_t stream) {
  const float* x      = (const float*)d_in[0];
  const int*   ei     = (const int*)  d_in[1];
  const float* ea     = (const float*)d_in[2];
  const float* pre_w  = (const float*)d_in[4];
  const float* pre_b  = (const float*)d_in[5];
  const float* gnp_w  = (const float*)d_in[6];
  const float* gnp_b  = (const float*)d_in[7];
  const float* gnp_ms = (const float*)d_in[8];
  const float* init_w = (const float*)d_in[9];
  const float* init_b = (const float*)d_in[10];
  const float* gni_w  = (const float*)d_in[11];
  const float* gni_b  = (const float*)d_in[12];
  const float* gni_ms = (const float*)d_in[13];
  const float* gw1    = (const float*)d_in[14];
  const float* gb1    = (const float*)d_in[15];
  const float* gw2    = (const float*)d_in[16];
  const float* gb2    = (const float*)d_in[17];
  const float* gw3    = (const float*)d_in[18];
  const float* gb3    = (const float*)d_in[19];
  const float* zw     = (const float*)d_in[20];
  const float* zb     = (const float*)d_in[21];
  const float* rw     = (const float*)d_in[22];
  const float* rb     = (const float*)d_in[23];
  const float* hw     = (const float*)d_in[24];
  const float* hb     = (const float*)d_in[25];
  const float* gng_w  = (const float*)d_in[26];
  const float* gng_b  = (const float*)d_in[27];
  const float* gng_ms = (const float*)d_in[28];
  const float* fc1w   = (const float*)d_in[29];
  const float* fc1b   = (const float*)d_in[30];
  const float* gnf_w  = (const float*)d_in[31];
  const float* gnf_b  = (const float*)d_in[32];
  const float* gnf_ms = (const float*)d_in[33];
  const float* fc2w   = (const float*)d_in[34];
  const float* fc2b   = (const float*)d_in[35];

  const int N = in_sizes[3];        // batch has N entries
  const int E = in_sizes[2];        // edge_attr is [E,1]
  float* out = (float*)d_out;

  // ---- workspace layout
  char* ws = (char*)d_ws;
  size_t off = 0;
  auto carve = [&](size_t bytes) { size_t o = off; off = (off + bytes + 255) & ~(size_t)255; return o; };
  float* H    = (float*)(ws + carve((size_t)N * 64 * 4));
  f16*   H16  = (f16*)  (ws + carve((size_t)N * 64 * 2));
  float* H0   = (float*)(ws + carve((size_t)N * 64 * 4));
  f16*   H016 = (f16*)  (ws + carve((size_t)N * 64 * 2));
  float* AGGR = (float*)(ws + carve((size_t)N * 64 * 4));
  float* HPRE = (float*)(ws + carve((size_t)N * 32 * 4));
  float* DIS  = (float*)(ws + carve((size_t)N * 4));
  float* GNS  = (float*)(ws + carve(2 * 64 * 4));
  f16* B1a  = (f16*)(ws + carve(8  * 512 * 2));
  f16* B1b  = (f16*)(ws + carve(8  * 512 * 2));
  f16* B2   = (f16*)(ws + carve(8  * 512 * 2));
  f16* Bz   = (f16*)(ws + carve(16 * 512 * 2));
  f16* Br   = (f16*)(ws + carve(16 * 512 * 2));
  f16* Bh   = (f16*)(ws + carve(16 * 512 * 2));
  f16* Bf1  = (f16*)(ws + carve(16 * 512 * 2));
  (void)ws_size; (void)n_in; (void)out_size;

  const float* w1e = gw1 + 128 * 64;   // row 128 of gate_w1 (edge_attr weights)

  // ---- weight swizzle
  k_swizzle<<<1, 256, 0, stream>>>(gw1,            B1a, 8);
  k_swizzle<<<1, 256, 0, stream>>>(gw1 + 64 * 64,  B1b, 8);
  k_swizzle<<<1, 256, 0, stream>>>(gw2,            B2,  8);
  k_swizzle<<<2, 256, 0, stream>>>(zw,             Bz, 16);
  k_swizzle<<<2, 256, 0, stream>>>(rw,             Br, 16);
  k_swizzle<<<2, 256, 0, stream>>>(hw,             Bh, 16);
  k_swizzle<<<2, 256, 0, stream>>>(fc1w,           Bf1, 16);

  // ---- degree -> dis
  hipMemsetAsync(DIS, 0, (size_t)N * 4, stream);
  k_deg<<<cdiv(E, 256), 256, 0, stream>>>(ei, DIS, E);
  k_dis<<<cdiv(N, 256), 256, 0, stream>>>(DIS, N);

  // ---- preproc + graphnorm + relu
  k_pre<<<cdiv(N * 32, 256), 256, 0, stream>>>(x, pre_w, pre_b, HPRE, N);
  hipMemsetAsync(GNS, 0, 2 * 32 * 4, stream);
  k_gn_reduce<<<64, 256, 0, stream>>>(HPRE, GNS, N, 32);
  k_gn_apply<<<cdiv(N * 32, 256), 256, 0, stream>>>(HPRE, GNS, gnp_w, gnp_b, gnp_ms,
                                                    HPRE, (f16*)nullptr, N, 32, 1);

  // ---- init + graphnorm + relu -> h0 (f32 + f16), h = h0
  k_init<<<cdiv(N * 64, 256), 256, 0, stream>>>(HPRE, init_w, init_b, H0, N);
  hipMemsetAsync(GNS, 0, 2 * 64 * 4, stream);
  k_gn_reduce<<<64, 256, 0, stream>>>(H0, GNS, N, 64);
  k_gn_apply<<<cdiv(N * 64, 256), 256, 0, stream>>>(H0, GNS, gni_w, gni_b, gni_ms,
                                                    H0, H016, N, 64, 1);
  hipMemcpyAsync(H,   H0,   (size_t)N * 64 * 4, hipMemcpyDeviceToDevice, stream);
  hipMemcpyAsync(H16, H016, (size_t)N * 64 * 2, hipMemcpyDeviceToDevice, stream);

  // ---- 4 tied GGNN layers
  const int etiles = (E + N) / 16;
  const int ntiles = N / 16;
  for (int layer = 0; layer < 4; ++layer) {
    hipMemsetAsync(AGGR, 0, (size_t)N * 64 * 4, stream);
    k_edge<<<cdiv(etiles, WPB), 256, 0, stream>>>(ei, ea, DIS, H16, AGGR,
                                                  B1a, B1b, B2, w1e, gb1, gb2, gw3, gb3,
                                                  E, N);
    k_node<<<cdiv(ntiles, WPB), 256, 0, stream>>>(H16, H, AGGR, Bz, Br, Bh,
                                                  zb, rb, hb, N);
    hipMemsetAsync(GNS, 0, 2 * 64 * 4, stream);
    k_gn_reduce<<<64, 256, 0, stream>>>(AGGR, GNS, N, 64);
    k_gn_apply<<<cdiv(N * 64, 256), 256, 0, stream>>>(AGGR, GNS, gng_w, gng_b, gng_ms,
                                                      H, H16, N, 64, 0);
  }

  // ---- fc1([h0,h]) + graphnorm + relu -> H
  k_fc1<<<cdiv(ntiles, WPB), 256, 0, stream>>>(H016, H16, AGGR, Bf1, fc1b, N);
  hipMemsetAsync(GNS, 0, 2 * 64 * 4, stream);
  k_gn_reduce<<<64, 256, 0, stream>>>(AGGR, GNS, N, 64);
  k_gn_apply<<<cdiv(N * 64, 256), 256, 0, stream>>>(AGGR, GNS, gnf_w, gnf_b, gnf_ms,
                                                    H, (f16*)nullptr, N, 64, 1);

  // ---- fc2 -> logits
  k_fc2<<<cdiv(N, 256), 256, 0, stream>>>(H, fc2w, fc2b, out, N);
}
